// PAMDec_3d_26749056320309
// MI455X (gfx1250) — compile-verified, matrix operand order checked by reading, not run on hardware
//
#include <hip/hip_runtime.h>

// ---------------------------------------------------------------------------
// PAMDec 3D attention, fused bf16-WMMA + TDM implementation for gfx1250
//   N=4, C=512, D=128, K=256, S = T*W*H = 8192
// ---------------------------------------------------------------------------

#define C_DIM 512
#define D_DIM 128
#define K_CTR 256
#define S_DIM 8192
#define N_BAT 4
#define BS    128        // S rows per workgroup
#define NTHR  256        // 8 wave32

typedef __bf16 bf16_t;
typedef unsigned int u32;
typedef __attribute__((ext_vector_type(4)))  __bf16 v4bf;
typedef __attribute__((ext_vector_type(8)))  __bf16 v8bf;
typedef __attribute__((ext_vector_type(16))) __bf16 v16bf;
typedef __attribute__((ext_vector_type(8)))  float  v8f;
typedef __attribute__((ext_vector_type(4)))  u32    v4u;
typedef __attribute__((ext_vector_type(8)))  u32    v8u;

__device__ __forceinline__ v8f vzero() {
  v8f z = {0.f, 0.f, 0.f, 0.f, 0.f, 0.f, 0.f, 0.f};
  return z;
}

__device__ __forceinline__ v4bf pack4(float a, float b, float c, float d) {
  v4bf p;
  p[0] = (bf16_t)a; p[1] = (bf16_t)b; p[2] = (bf16_t)c; p[3] = (bf16_t)d;
  return p;
}

// Fragment load for 16-bit A/B of v_wmma_*_16x16x32 (see ISA 7.12.2):
//   lane L: idx = L&15 (row for A / col for B), koff = (L>>4)*8
//   halves [0..7]  = M[idx][kb + koff + i]
//   halves [8..15] = M[idx][kb + 16 + koff + i]
__device__ __forceinline__ v16bf frag_ld(const bf16_t* p, int koff) {
  v8bf lo = *(const v8bf*)(p + koff);
  v8bf hi = *(const v8bf*)(p + 16 + koff);
  return __builtin_shufflevector(lo, hi, 0, 1, 2, 3, 4, 5, 6, 7,
                                 8, 9, 10, 11, 12, 13, 14, 15);
}

__device__ __forceinline__ v8f wmma_bf16(v16bf a, v16bf b, v8f c) {
  return __builtin_amdgcn_wmma_f32_16x16x32_bf16(false, a, false, b,
                                                 (short)0, c, false, false);
}

// ---------------------------------------------------------------------------
// TDM: async 2D bf16 tile load Global -> LDS with row padding.
// D# group0: count=1 | lds_addr | global_addr[56:0] | type=2.
// D# group1: data_size=1(2B), pad_enable, pad_interval P (pad every 2^(P+1)
//            DWORDs), pad_amount A (A+1 DWORDs), tensor/tile dims, dim0 stride.
// Issue from one wave; completion tracked with TENSORcnt.
// ---------------------------------------------------------------------------
__device__ __forceinline__ void tdm_load_2d_bf16(u32 lds_byte_off,
                                                 const void* gaddr, u32 cols,
                                                 u32 rows, u32 stride_elems,
                                                 u32 pad_interval,
                                                 u32 pad_amount) {
  unsigned long long ga = (unsigned long long)gaddr;
  v4u g0;
  g0[0] = 1u;                                             // count=1
  g0[1] = lds_byte_off;                                   // lds_addr
  g0[2] = (u32)ga;                                        // global_addr[31:0]
  g0[3] = (u32)((ga >> 32) & 0x1FFFFFFu) | 0x80000000u;   // [56:32] | type=2
  v8u g1;
  g1[0] = (1u << 16)                                      // data_size = 2B
        | (1u << 20)                                      // pad_enable
        | (pad_interval << 22) | (pad_amount << 25);
  g1[1] = (cols & 0xFFFFu) << 16;                         // tensor_dim0[15:0]
  g1[2] = (rows & 0xFFFFu) << 16;   // tensor_dim0[31:16]=0 | tensor_dim1[15:0]
  g1[3] = (cols & 0xFFFFu) << 16;   // tensor_dim1[31:16]=0 | tile_dim0
  g1[4] = rows & 0xFFFFu;           // tile_dim1 (tile_dim2 = 0)
  g1[5] = stride_elems;             // tensor_dim0_stride[31:0]
  g1[6] = 0u;
  g1[7] = 0u;
  asm volatile("tensor_load_to_lds %0, %1" ::"s"(g0), "s"(g1) : "memory");
}

// ---------------------------------------------------------------------------
// Prep: Out[i][j] = sum_c A[i][c]*B[j][c] + bias  (fp32 in -> bf16 out)
// k = y*Wk^T + bk (bias cols) ; v^T = Wv*y^T + bv (bias rows)
// ---------------------------------------------------------------------------
__global__ __launch_bounds__(NTHR) void prep_gemm(
    const float* __restrict__ A, int lda, long long strideA,
    const float* __restrict__ B, int ldb, long long strideB,
    const float* __restrict__ bias, int bias_rows,
    bf16_t* __restrict__ Out, int ldo, long long strideO) {
  __shared__ bf16_t as[128 * 40];
  __shared__ bf16_t bs[128 * 40];

  const int bz = blockIdx.z;
  A += (long long)bz * strideA;
  B += (long long)bz * strideB;
  Out += (long long)bz * strideO;

  const int ib = blockIdx.x * 128, jb = blockIdx.y * 128;
  const int tid = threadIdx.x, wave = tid >> 5, lane = tid & 31;
  const int lrow = lane & 15, hi = lane >> 4, koff = hi * 8;
  const int arow = wave * 16 + lrow;

  v8f acc[8];
#pragma unroll
  for (int t = 0; t < 8; ++t) acc[t] = vzero();

  for (int cb = 0; cb < C_DIM; cb += 32) {
    __syncthreads();
#pragma unroll
    for (int i = 0; i < 4; ++i) {  // 128x32 stage via b128 + packed bf16
      int idx = i * NTHR + tid;
      int r = idx >> 3, c4 = (idx & 7) * 4;
      float4 fa = *(const float4*)&A[(long long)(ib + r) * lda + cb + c4];
      float4 fb = *(const float4*)&B[(long long)(jb + r) * ldb + cb + c4];
      *(v4bf*)&as[r * 40 + c4] = pack4(fa.x, fa.y, fa.z, fa.w);
      *(v4bf*)&bs[r * 40 + c4] = pack4(fb.x, fb.y, fb.z, fb.w);
    }
    __syncthreads();
    v16bf a = frag_ld(&as[arow * 40], koff);
#pragma unroll
    for (int t = 0; t < 8; ++t) {
      v16bf b = frag_ld(&bs[(t * 16 + lrow) * 40], koff);
      acc[t] = wmma_bf16(a, b, acc[t]);
    }
  }

#pragma unroll
  for (int t = 0; t < 8; ++t)
#pragma unroll
    for (int j = 0; j < 8; ++j) {
      int r = wave * 16 + hi * 8 + j;   // C/D layout: row = j + 8*(lane>>4)
      int c = t * 16 + lrow;            //             col = lane&15
      float v = acc[t][j] + (bias_rows ? bias[ib + r] : bias[jb + c]);
      Out[(long long)(ib + r) * ldo + jb + c] = (bf16_t)v;
    }
}

// ---------------------------------------------------------------------------
// Fused kernel. Dynamic LDS (bytes):
//   [      0, 34816) q_lds [128][136] bf16     (phase C: o_lds [128][132] f32,
//   [  34816,104448) k_lds [256][136] bf16      spans dead q+k regions)
//   [ 104448,172032) at_lds [128][264] bf16    (phase A: x/Wq staging)
//   [ 172032,239616) v_lds0 [128][264] bf16    (TDM double buffer)
//   [ 239616,307200) v_lds1 [128][264] bf16
// k, v chunks arrive via TDM issued at kernel entry, overlapping phases A/B.
// ---------------------------------------------------------------------------
#define SMEM_Q   0
#define SMEM_K   34816
#define SMEM_AT  104448
#define SMEM_V0  172032
#define SMEM_V1  239616
#define SMEM_TOT 307200

__global__ __launch_bounds__(NTHR) void pam_fused(
    const float* __restrict__ x, const float* __restrict__ Wq,
    const float* __restrict__ bq, const bf16_t* __restrict__ kmat,
    const bf16_t* __restrict__ vmat, const float* __restrict__ scale,
    float* __restrict__ out) {
  extern __shared__ char smem[];
  bf16_t* q_lds = (bf16_t*)(smem + SMEM_Q);     // [128][136]
  bf16_t* k_lds = (bf16_t*)(smem + SMEM_K);     // [256][136]
  bf16_t* at_lds = (bf16_t*)(smem + SMEM_AT);   // [128][264]
  bf16_t* v_lds0 = (bf16_t*)(smem + SMEM_V0);   // [128][264]
  bf16_t* v_lds1 = (bf16_t*)(smem + SMEM_V1);   // [128][264]
  float* o_lds = (float*)(smem + SMEM_Q);       // [128][132] f32 (phase C)
  bf16_t* xs = at_lds;                          // [128][40] phase-A staging
  bf16_t* wqs = at_lds + 5120;                  // [128][40]

  const int n = blockIdx.y;
  const int s0 = blockIdx.x * BS;
  const int tid = threadIdx.x, wave = tid >> 5, lane = tid & 31;
  const int lrow = lane & 15, hi = lane >> 4, koff = hi * 8;
  const int arow = wave * 16 + lrow;

  const float* xn = x + (long long)n * C_DIM * S_DIM;
  const bf16_t* kn = kmat + (long long)n * K_CTR * D_DIM;
  const bf16_t* vn = vmat + (long long)n * C_DIM * K_CTR;  // v^T: [512][256]
  const u32 lds_base = (u32)__builtin_amdgcn_groupstaticsize();

  // Kick off TDM DMA: k (256x128, pad 256B->+16B) and first two v^T chunks
  // (128x256, pad 512B->+16B). These overlap the whole Q/energy computation.
  if (tid < 32) {
    tdm_load_2d_bf16(lds_base + SMEM_K, kn, D_DIM, K_CTR, D_DIM, 5u, 3u);
    tdm_load_2d_bf16(lds_base + SMEM_V0, vn, K_CTR, 128, K_CTR, 6u, 3u);
    tdm_load_2d_bf16(lds_base + SMEM_V1, vn + 128 * K_CTR, K_CTR, 128, K_CTR,
                     6u, 3u);
  }

  // ---------------- Phase A: Q[128][128] = X_tile^T * Wq^T + bq -------------
  v8f accQ[8];
#pragma unroll
  for (int t = 0; t < 8; ++t) accQ[t] = vzero();

  for (int cb = 0; cb < C_DIM; cb += 32) {
    __syncthreads();
#pragma unroll
    for (int i = 0; i < 4; ++i) {  // x: C-major -> xs[s][c] bf16, b128 loads
      int idx = i * NTHR + tid;
      int c = idx >> 5, s4 = (idx & 31) * 4;
      const float4* gp = (const float4*)&xn[(long long)(cb + c) * S_DIM + s0 + s4];
      float4 f = *gp;
      xs[(s4 + 0) * 40 + c] = (bf16_t)f.x;
      xs[(s4 + 1) * 40 + c] = (bf16_t)f.y;
      xs[(s4 + 2) * 40 + c] = (bf16_t)f.z;
      xs[(s4 + 3) * 40 + c] = (bf16_t)f.w;
      if (cb + 32 < C_DIM) __builtin_prefetch(gp + 8 * S_DIM, 0, 1);
    }
#pragma unroll
    for (int i = 0; i < 4; ++i) {  // Wq[d][c] -> wqs[d][c], packed b64 stores
      int idx = i * NTHR + tid;
      int d = idx >> 3, c4 = (idx & 7) * 4;
      float4 f = *(const float4*)&Wq[d * C_DIM + cb + c4];
      *(v4bf*)&wqs[d * 40 + c4] = pack4(f.x, f.y, f.z, f.w);
    }
    __syncthreads();
    v16bf a = frag_ld(&xs[arow * 40], koff);
#pragma unroll
    for (int t = 0; t < 8; ++t) {
      v16bf b = frag_ld(&wqs[(t * 16 + lrow) * 40], koff);
      accQ[t] = wmma_bf16(a, b, accQ[t]);
    }
  }
#pragma unroll
  for (int t = 0; t < 8; ++t)
#pragma unroll
    for (int j = 0; j < 8; ++j) {
      int r = wave * 16 + hi * 8 + j;
      int d = t * 16 + lrow;
      q_lds[r * 136 + d] = (bf16_t)(accQ[t][j] + bq[d]);
    }
  if (tid < 32) __builtin_amdgcn_s_wait_tensorcnt(2);  // k DMA complete
  __syncthreads();

  // ---------------- Phase B: energy = Q * K^T, softmax over 256 -------------
  v8f accE[16];
#pragma unroll
  for (int t = 0; t < 16; ++t) accE[t] = vzero();
#pragma unroll
  for (int db = 0; db < D_DIM; db += 32) {
    v16bf a = frag_ld(&q_lds[arow * 136 + db], koff);
#pragma unroll
    for (int t = 0; t < 16; ++t) {
      v16bf b = frag_ld(&k_lds[(t * 16 + lrow) * 136 + db], koff);
      accE[t] = wmma_bf16(a, b, accE[t]);
    }
  }

  // softmax: row r's 256 values sit in register j = r&7 across one 16-lane
  // half-wave (C/D layout) x 16 tiles -> reduce over t, then shfl_xor
#pragma unroll
  for (int j = 0; j < 8; ++j) {
    float m = -3.0e38f;
#pragma unroll
    for (int t = 0; t < 16; ++t) m = fmaxf(m, accE[t][j]);
    for (int off = 1; off < 16; off <<= 1) m = fmaxf(m, __shfl_xor(m, off, 32));
    float ssum = 0.f;
#pragma unroll
    for (int t = 0; t < 16; ++t) {
      float p = __expf(accE[t][j] - m);
      accE[t][j] = p;
      ssum += p;
    }
    for (int off = 1; off < 16; off <<= 1) ssum += __shfl_xor(ssum, off, 32);
    float rinv = 1.f / ssum;
    int r = wave * 16 + hi * 8 + j;
#pragma unroll
    for (int t = 0; t < 16; ++t)
      at_lds[r * 264 + t * 16 + lrow] = (bf16_t)(accE[t][j] * rinv);
  }
  __syncthreads();

  // ---------------- Phase C: out = attn * V (+ scale, + x) ------------------
  const float sc = scale[0];
  float* outn = out + (long long)n * C_DIM * S_DIM;

#pragma unroll
  for (int i4 = 0; i4 < 4; ++i4) {
    const int ob = i4 * 128;
    bf16_t* vb = (i4 & 1) ? v_lds1 : v_lds0;

    if (tid < 32) {  // wave0 issued k,v0,v1,(v2),(v3) in order
      if (i4 < 3) __builtin_amdgcn_s_wait_tensorcnt(1);
      else        __builtin_amdgcn_s_wait_tensorcnt(0);
    }
    __syncthreads();  // v chunk i4 resident for all waves

    v8f accO[8];
#pragma unroll
    for (int t = 0; t < 8; ++t) accO[t] = vzero();
#pragma unroll
    for (int kb = 0; kb < K_CTR; kb += 32) {
      v16bf a = frag_ld(&at_lds[arow * 264 + kb], koff);
#pragma unroll
      for (int t = 0; t < 8; ++t) {
        v16bf b = frag_ld(&vb[(t * 16 + lrow) * 264 + kb], koff);
        accO[t] = wmma_bf16(a, b, accO[t]);
      }
    }
    __syncthreads();  // everyone done reading this v buffer

    // prefetch chunk i4+2 into the buffer we just freed (async, overlaps
    // the transpose + epilogue + next chunk's GEMM)
    if (i4 + 2 < 4 && tid < 32)
      tdm_load_2d_bf16(lds_base + ((i4 & 1) ? SMEM_V1 : SMEM_V0),
                       vn + (i4 + 2) * 128 * K_CTR, K_CTR, 128, K_CTR, 6u, 3u);

    // transpose through LDS (reusing dead q/k region) for coalesced stores
#pragma unroll
    for (int t = 0; t < 8; ++t)
#pragma unroll
      for (int j = 0; j < 8; ++j) {
        int o = t * 16 + lrow;
        int s = wave * 16 + hi * 8 + j;
        o_lds[o * 132 + s] = accO[t][j];
      }
    __syncthreads();

#pragma unroll
    for (int i = 0; i < 16; ++i) {  // b128 epilogue; x re-read hits L2
      int idx = i * NTHR + tid;
      int o = idx >> 5, s4 = (idx & 31) * 4;
      long long xoff = (long long)(ob + o) * S_DIM + s0 + s4;
      float4 ov = *(const float4*)&o_lds[o * 132 + s4];
      float4 xv = *(const float4*)&xn[xoff];
      float4 r;
      r.x = sc * ov.x + xv.x;
      r.y = sc * ov.y + xv.y;
      r.z = sc * ov.z + xv.z;
      r.w = sc * ov.w + xv.w;
      *(float4*)&outn[xoff] = r;
    }
    __syncthreads();
  }
}

// ---------------------------------------------------------------------------
extern "C" void kernel_launch(void* const* d_in, const int* in_sizes, int n_in,
                              void* d_out, int out_size, void* d_ws,
                              size_t ws_size, hipStream_t stream) {
  const float* x = (const float*)d_in[0];
  const float* y = (const float*)d_in[1];
  const float* Wq = (const float*)d_in[2];
  const float* bq = (const float*)d_in[3];
  const float* Wk = (const float*)d_in[4];
  const float* bk = (const float*)d_in[5];
  const float* Wv = (const float*)d_in[6];
  const float* bv = (const float*)d_in[7];
  const float* scale = (const float*)d_in[8];
  float* out = (float*)d_out;

  // workspace: k bf16 (N,256,128) then v^T bf16 (N,512,256)
  bf16_t* k_ws = (bf16_t*)d_ws;
  bf16_t* v_ws = (bf16_t*)((char*)d_ws + (size_t)N_BAT * K_CTR * D_DIM * 2);

  dim3 blk(NTHR);

  // k[n] = y[n] * Wk^T + bk  -> (256 x 128)
  prep_gemm<<<dim3(2, 1, N_BAT), blk, 0, stream>>>(
      y, C_DIM, (long long)K_CTR * C_DIM, Wk, C_DIM, 0LL, bk, 0, k_ws, D_DIM,
      (long long)K_CTR * D_DIM);

  // v^T[n] = Wv * y[n]^T + bv -> (512 x 256)
  prep_gemm<<<dim3(4, 2, N_BAT), blk, 0, stream>>>(
      Wv, C_DIM, 0LL, y, C_DIM, (long long)K_CTR * C_DIM, bv, 1, v_ws, K_CTR,
      (long long)C_DIM * K_CTR);

  hipFuncSetAttribute((const void*)pam_fused,
                      hipFuncAttributeMaxDynamicSharedMemorySize, SMEM_TOT);
  pam_fused<<<dim3(S_DIM / BS, N_BAT), blk, SMEM_TOT, stream>>>(
      x, Wq, bq, k_ws, v_ws, scale, out);
}
